// Quantizer_31988916420863
// MI455X (gfx1250) — matrix operand order repaired and checked by reading, not compiled, not run on hardware
//
#include <hip/hip_runtime.h>
#include <hip/hip_bf16.h>

typedef __attribute__((ext_vector_type(16))) __bf16 v16bf;
typedef __attribute__((ext_vector_type(8)))  __bf16 v8bf;
typedef __attribute__((ext_vector_type(4)))  __bf16 v4bf;
typedef __attribute__((ext_vector_type(8)))  float  v8f;

#define M_TOTAL   (8 * 2048)     // B*T = 16384 rows of x
#define D_DIM     256            // feature dim (GEMM K)
#define K_CB      8192           // codebook entries (GEMM N)
#define N_TILE    64             // codebook rows staged in LDS per tile
#define N_TILES   (K_CB / N_TILE)
#define LDS_STRIDE 264           // bf16 elems per LDS row (528B row -> conflict-free b128)
#define D_STEPS   (D_DIM / 32)   // 8 WMMA K-steps of 32
#define MT        2              // M-tiles (16 rows each) per wave
#define M_BLOCK   (8 * 16 * MT)  // 256 rows per 8-wave block

union BF16x16 { v16bf v; v8bf h[2]; };

// ---------------- fp32 -> bf16 conversion of x ----------------
__global__ __launch_bounds__(256) void vq_convert_x(const float* __restrict__ xs,
                                                    __bf16* __restrict__ xb) {
  int i = (blockIdx.x * 256 + threadIdx.x) * 4;
  float4 f = *(const float4*)(xs + i);
  v4bf o;
  o[0] = (__bf16)f.x; o[1] = (__bf16)f.y; o[2] = (__bf16)f.z; o[3] = (__bf16)f.w;
  *(v4bf*)(xb + i) = o;
}

// ------- fp32 -> bf16 conversion of codebook + 0.5*||c||^2 -------
__global__ __launch_bounds__(256) void vq_convert_cb(const float* __restrict__ cb,
                                                     __bf16* __restrict__ cbb,
                                                     float* __restrict__ csq) {
  int row = blockIdx.x, t = threadIdx.x;
  float v = cb[row * D_DIM + t];
  cbb[row * D_DIM + t] = (__bf16)v;
  float s = v * v;
  #pragma unroll
  for (int m = 16; m >= 1; m >>= 1) s += __shfl_xor(s, m, 32);
  __shared__ float red[8];
  if ((t & 31) == 0) red[t >> 5] = s;
  __syncthreads();
  if (t == 0) {
    float tot = 0.f;
    #pragma unroll
    for (int w = 0; w < 8; ++w) tot += red[w];
    csq[row] = 0.5f * tot;
  }
}

// Async-copy one 64-row codebook tile (32KB bf16) global -> LDS.
// Per thread: 8 chunks of 16B; each asm stmt is one wave-level
// GLOBAL_LOAD_ASYNC_TO_LDS_B128 moving 512B with per-lane addresses (ASYNCcnt).
__device__ __forceinline__ void async_stage_tile(const __bf16* __restrict__ cbb, int tt,
                                                 __bf16* dstbase, int tid) {
  const char* gsrc = (const char*)(cbb + (size_t)tt * N_TILE * D_DIM);
  #pragma unroll
  for (int i = 0; i < 8; ++i) {
    int c    = tid + i * 256;            // 2048 chunks of 16B
    int row  = c >> 5;                   // 32 chunks per 512B row
    int offb = (c & 31) * 16;            // byte offset within row
    // Flat->LDS mapping truncates to addr[31:0]; low 32 bits of a shared
    // pointer are a valid LDS byte address for the async op's VDST.
    unsigned lds_addr = (unsigned)(uintptr_t)(dstbase + row * LDS_STRIDE) + (unsigned)offb;
    unsigned long long gaddr = (unsigned long long)(uintptr_t)(gsrc + (size_t)c * 16);
    asm volatile("global_load_async_to_lds_b128 %0, %1, off"
                 :: "v"(lds_addr), "v"(gaddr) : "memory");
  }
}

// ---------------- fused GEMM (bf16 WMMA) + argmin ----------------
// Block: 256 threads = 8 waves; wave w owns 32 M rows (2 WMMA M-tiles).
// dist_rel(k) = 0.5*||c_k||^2 - x . c_k   (same argmin as full euclidean dist)
__global__ __launch_bounds__(256) void vq_argmin(const __bf16* __restrict__ xb,
                                                 const __bf16* __restrict__ cbb,
                                                 const float* __restrict__ csq,
                                                 int* __restrict__ idx_out) {
  __shared__ __bf16 lds[2][N_TILE * LDS_STRIDE];   // 2 x 33792 B
  const int tid  = threadIdx.x;
  const int wave = tid >> 5;
  const int lane = tid & 31;
  const int l16  = lane & 15;
  const int hi   = lane >> 4;
  const int mbase = blockIdx.x * M_BLOCK + wave * (16 * MT);

  // A fragments: MT tiles of (16 rows x 256 K), ISA 16-bit A layout:
  // lane<16: K {d0+0..7, d0+16..23}; lane>=16: K {d0+8..15, d0+24..31}
  BF16x16 a[MT][D_STEPS];
  #pragma unroll
  for (int mt = 0; mt < MT; ++mt) {
    const __bf16* xrow = xb + (size_t)(mbase + mt * 16 + l16) * D_DIM;
    #pragma unroll
    for (int s = 0; s < D_STEPS; ++s) {
      int d0 = s * 32;
      a[mt][s].h[0] = *(const v8bf*)(xrow + d0 + hi * 8);
      a[mt][s].h[1] = *(const v8bf*)(xrow + d0 + 16 + hi * 8);
    }
  }

  float minv[MT][8];
  int   mini[MT][8];
  #pragma unroll
  for (int mt = 0; mt < MT; ++mt)
    #pragma unroll
    for (int j = 0; j < 8; ++j) { minv[mt][j] = 3.4e38f; mini[mt][j] = 0; }

  // Prologue: async-stage tile 0 into lds[0]
  async_stage_tile(cbb, 0, &lds[0][0], tid);
  asm volatile("s_wait_asynccnt 0" ::: "memory");
  __syncthreads();

  const v8f vzero = {0.f, 0.f, 0.f, 0.f, 0.f, 0.f, 0.f, 0.f};

  for (int t = 0; t < N_TILES; ++t) {
    const int buf = t & 1;

    // Kick off async copy of tile t+1 into the other LDS buffer; it stays
    // in flight across the whole WMMA block below (ASYNCcnt tracked).
    if (t + 1 < N_TILES)
      async_stage_tile(cbb, t + 1, &lds[buf ^ 1][0], tid);
    // Pull tile t+2 toward the L2/WGP: one 128B line per thread covers 32KB
    if (t + 2 < N_TILES)
      __builtin_prefetch(cbb + (size_t)(t + 2) * N_TILE * D_DIM + tid * 64, 0, 0);

    // B layout: column n = l16; lane<16 holds K {d0+0..7, d0+16..23}, lane>=16 K {d0+8..15, d0+24..31}
    const __bf16* base = &lds[buf][0] + l16 * LDS_STRIDE + hi * 8;

    v8f acc[MT][4];
    #pragma unroll
    for (int mt = 0; mt < MT; ++mt)
      #pragma unroll
      for (int ns = 0; ns < 4; ++ns) acc[mt][ns] = vzero;

    // Software-pipelined B fragments: double register buffer over D-steps.
    // Each B fragment feeds MT WMMAs -> ds:wmma = 1:1.
    BF16x16 b[2][4];
    #pragma unroll
    for (int ns = 0; ns < 4; ++ns) {
      const __bf16* p = base + ns * 16 * LDS_STRIDE;
      b[0][ns].h[0] = *(const v8bf*)(p);
      b[0][ns].h[1] = *(const v8bf*)(p + 16);
    }
    #pragma unroll
    for (int s = 0; s < D_STEPS; ++s) {
      const int cur = s & 1, nxt = cur ^ 1;
      if (s + 1 < D_STEPS) {
        #pragma unroll
        for (int ns = 0; ns < 4; ++ns) {
          const __bf16* p = base + ns * 16 * LDS_STRIDE + (s + 1) * 32;
          b[nxt][ns].h[0] = *(const v8bf*)(p);
          b[nxt][ns].h[1] = *(const v8bf*)(p + 16);
        }
      }
      // 8 independent accumulator chains per D-step -> no WMMA RAW hazard NOPs
      #pragma unroll
      for (int mt = 0; mt < MT; ++mt)
        #pragma unroll
        for (int ns = 0; ns < 4; ++ns)
          acc[mt][ns] = __builtin_amdgcn_wmma_f32_16x16x32_bf16(
              false, a[mt][s].v, false, b[cur][ns].v, (short)0, acc[mt][ns], false, false);
    }

    // argmin update (columns interleave mod 16 across lanes)
    #pragma unroll
    for (int ns = 0; ns < 4; ++ns) {
      int ncol = t * N_TILE + ns * 16 + l16;
      float cs = csq[ncol];
      #pragma unroll
      for (int mt = 0; mt < MT; ++mt)
        #pragma unroll
        for (int j = 0; j < 8; ++j) {
          float d = cs - acc[mt][ns][j];
          if (d < minv[mt][j]) { minv[mt][j] = d; mini[mt][j] = ncol; }
        }
    }

    // Drain my async copy of tile t+1, then one barrier per tile
    if (t + 1 < N_TILES) {
      asm volatile("s_wait_asynccnt 0" ::: "memory");
      __syncthreads();
    }
  }

  // Min-reduce across the 16 lanes of each half-wave (xor keeps bit 4 fixed).
  #pragma unroll
  for (int off = 8; off >= 1; off >>= 1) {
    #pragma unroll
    for (int mt = 0; mt < MT; ++mt)
      #pragma unroll
      for (int j = 0; j < 8; ++j) {
        float ov = __shfl_xor(minv[mt][j], off, 32);
        int   oi = __shfl_xor(mini[mt][j], off, 32);
        if (ov < minv[mt][j] || (ov == minv[mt][j] && oi < mini[mt][j])) {
          minv[mt][j] = ov; mini[mt][j] = oi;
        }
      }
  }
  if (l16 == 0) {                       // lane 0 -> rows 0-7, lane 16 -> rows 8-15
    #pragma unroll
    for (int mt = 0; mt < MT; ++mt)
      #pragma unroll
      for (int j = 0; j < 8; ++j)
        idx_out[mbase + mt * 16 + hi * 8 + j] = mini[mt][j];
  }
}

// ------- exact fp32 commitment loss: sum (cb[idx[m]][d]-x[m][d])^2 -------
__global__ __launch_bounds__(256) void vq_commit(const float* __restrict__ xs,
                                                 const float* __restrict__ cb,
                                                 const int* __restrict__ idx,
                                                 float* __restrict__ accum) {
  int t = threadIdx.x;                  // d index
  int mbase = blockIdx.x * 8;
  float s = 0.f;
  #pragma unroll
  for (int r = 0; r < 8; ++r) {
    int m = mbase + r;
    int k = idx[m];
    float d = cb[(size_t)k * D_DIM + t] - xs[(size_t)m * D_DIM + t];
    s += d * d;
  }
  #pragma unroll
  for (int msk = 16; msk >= 1; msk >>= 1) s += __shfl_xor(s, msk, 32);
  __shared__ float red[8];
  if ((t & 31) == 0) red[t >> 5] = s;
  __syncthreads();
  if (t == 0) {
    float tot = 0.f;
    #pragma unroll
    for (int w = 0; w < 8; ++w) tot += red[w];
    atomicAdd(accum, tot);
  }
}

__global__ void vq_zero(float* accum) { *accum = 0.f; }

__global__ void vq_finalize(const float* __restrict__ accum, float* __restrict__ out) {
  float commit = *accum * (1.0f / (float)(M_TOTAL * D_DIM));
  out[0] = 0.25f * commit;   // loss
  out[1] = commit;           // commit_loss
}

extern "C" void kernel_launch(void* const* d_in, const int* in_sizes, int n_in,
                              void* d_out, int out_size, void* d_ws, size_t ws_size,
                              hipStream_t stream) {
  const float* xs = (const float*)d_in[0];   // (8,2048,256) f32
  // d_in[1] = ilens (int64), all == T -> slice is a no-op, unused
  const float* cb = (const float*)d_in[2];   // (8192,256) f32
  float* out = (float*)d_out;                // [loss, commit_loss]

  char* ws = (char*)d_ws;
  __bf16* xb    = (__bf16*)ws;                                          // 8 MiB
  __bf16* cbb   = (__bf16*)(ws + (size_t)8 * 1024 * 1024);              // 4 MiB
  float*  csq   = (float*) (ws + (size_t)12 * 1024 * 1024);             // 32 KiB
  int*    idx   = (int*)   (ws + (size_t)12 * 1024 * 1024 + 32 * 1024); // 64 KiB
  float*  accum = (float*) (ws + (size_t)12 * 1024 * 1024 + 96 * 1024); // 4 B

  vq_convert_x <<<(M_TOTAL * D_DIM) / (256 * 4), 256, 0, stream>>>(xs, xb);
  vq_convert_cb<<<K_CB, 256, 0, stream>>>(cb, cbb, csq);
  vq_zero      <<<1, 1, 0, stream>>>(accum);
  vq_argmin    <<<M_TOTAL / M_BLOCK, 256, 0, stream>>>(xb, cbb, csq, idx);
  vq_commit    <<<M_TOTAL / 8, 256, 0, stream>>>(xs, cb, idx, accum);
  vq_finalize  <<<1, 1, 0, stream>>>(accum, out);
}
